// SingleView_19318762898072
// MI455X (gfx1250) — matrix operand order repaired and checked by reading, not compiled, not run on hardware
//
#include <hip/hip_runtime.h>
#include <hip/hip_bf16.h>

typedef __attribute__((ext_vector_type(16))) _Float16 v16h;
typedef __attribute__((ext_vector_type(8)))  _Float16 v8h;
typedef __attribute__((ext_vector_type(8)))  float    v8f;
typedef __attribute__((ext_vector_type(4)))  unsigned int u32x4;
typedef __attribute__((ext_vector_type(8)))  int i32x8;
typedef __attribute__((ext_vector_type(4)))  int i32x4;

#define PC_MIN (-51.2f)
#define VOXEL  (0.4f)

// ---------------------------------------------------------------------------
// TDM: DMA a 2D tile (ph rows of td0 2-byte elements, global row stride
// `rowstride` elements) into contiguous LDS at lds_base. Issued by one wave;
// caller syncs the workgroup afterwards. D# fields per CDNA5 ISA.
// ---------------------------------------------------------------------------
__device__ __forceinline__ void tdm_load_patch(const _Float16* src, unsigned lds_base,
                                               int td0, int ph, int rowstride)
{
#if __has_builtin(__builtin_amdgcn_tensor_load_to_lds)
  unsigned long long ga = (unsigned long long)(const void*)src;
  u32x4 g0 = { 1u,                                   // count=1, user descriptor
               lds_base,                             // lds_addr (bytes)
               (unsigned)ga,                         // global_addr[31:0]
               (unsigned)((ga >> 32) & 0x01ffffffu)  // global_addr[56:32]
                   | (2u << 30) };                   // type=2 ("image")
  i32x8 g1 = { (int)0x00010000,                      // data_size=2B; no pad/iter
               (int)((td0 & 0xffff) << 16),          // tensor_dim0[15:0]
               (int)(((td0 >> 16) & 0xffff) | ((ph & 0xffff) << 16)), // td0 hi | td1 lo
               (int)((td0 & 0xffff) << 16),          // tile_dim0 (td1 hi = 0)
               (int)(ph & 0xffff),                   // tile_dim1 (tile_dim2=0)
               rowstride,                            // tensor_dim0_stride[31:0]
               0, 0 };
  i32x4 gz = {};
#if __clang_major__ >= 23
  i32x8 gz8 = {};
  __builtin_amdgcn_tensor_load_to_lds(g0, g1, gz, gz, gz8, 0);
#else
  __builtin_amdgcn_tensor_load_to_lds(g0, g1, gz, gz, 0);
#endif
  __builtin_amdgcn_s_wait_tensorcnt(0);
#endif
}

// ---------------------------------------------------------------------------
// PFN layer 0: per-point linear(16->16) + BN + ReLU; writes fp16 point feats
// and scatter-max (bitwise int atomicMax, valid for post-ReLU >= 0 values).
// ---------------------------------------------------------------------------
__global__ __launch_bounds__(256) void pfn0_kernel(
    const float* __restrict__ feat, const float* __restrict__ w0,
    const float* __restrict__ g0, const float* __restrict__ b0,
    const int* __restrict__ unq_inv, _Float16* __restrict__ pf16,
    float* __restrict__ hmax, int N)
{
  __shared__ float sw[256], sg[16], sb[16];
  int t = threadIdx.x;
  sw[t] = w0[t];
  if (t < 16) { sg[t] = g0[t]; sb[t] = b0[t]; }
  __syncthreads();
  int p = blockIdx.x * 256 + t;
  if (p >= N) return;
  float x[16];
#pragma unroll
  for (int j = 0; j < 16; ++j) x[j] = feat[(size_t)p * 16 + j];
  int piv = unq_inv[p];
#pragma unroll
  for (int c = 0; c < 16; ++c) {
    float s = 0.f;
#pragma unroll
    for (int j = 0; j < 16; ++j) s = fmaf(x[j], sw[j * 16 + c], s);
    s = fmaxf(s * sg[c] + sb[c], 0.f);
    pf16[(size_t)p * 32 + c] = (_Float16)s;
    atomicMax((int*)&hmax[(size_t)piv * 16 + c], __float_as_int(s));
  }
}

// append per-pillar max back to each point's feature vector (channels 16..31)
__global__ void concat_hmax_kernel(_Float16* __restrict__ pf16,
                                   const float* __restrict__ hmax,
                                   const int* __restrict__ unq_inv, int N)
{
  int p = blockIdx.x * blockDim.x + threadIdx.x;
  if (p >= N) return;
  int piv = unq_inv[p];
#pragma unroll
  for (int j = 0; j < 16; ++j)
    pf16[(size_t)p * 32 + 16 + j] = (_Float16)hmax[(size_t)piv * 16 + j];
}

// ---------------------------------------------------------------------------
// Weight repack into per-lane WMMA B-fragment layout:
// entry i = ((nt*taps + t)*KC + kc)*32 + lane holds 16 contiguous halfs:
// w[t*Cin + kc*32 + (lane>=16 ? 8 : 0) + (j<8 ? j : j+8)][nt*16 + lane%16], j = 0..15.
// ---------------------------------------------------------------------------
__global__ void pack_w_kernel(const float* __restrict__ w, _Float16* __restrict__ wp,
                              int Cin, int Cout, int taps)
{
  int KC = Cin / 32, NT = Cout / 16;
  int total = NT * taps * KC * 32;
  int i = blockIdx.x * blockDim.x + threadIdx.x;
  if (i >= total) return;
  int lane = i & 31;
  int r = i >> 5;
  int kc = r % KC; r /= KC;
  int t = r % taps; int nt = r / taps;
  int kb = kc * 32 + (lane >> 4) * 8;
  int n = nt * 16 + (lane & 15);
#pragma unroll
  for (int j = 0; j < 16; ++j)
    wp[(size_t)i * 16 + j] = (_Float16)w[(size_t)(t * Cin + kb + (j < 8 ? j : j + 8)) * Cout + n];
}

// ---------------------------------------------------------------------------
// PFN layer 1 via WMMA: [N,32] @ [32,64]. Each wave: 16 points x 64 channels,
// one A-fragment shared by 4 WMMAs. Epilogue: BN+ReLU, scatter-max into fv.
// Full tiles take an unguarded path (no per-lane exec juggling).
// ---------------------------------------------------------------------------
__global__ __launch_bounds__(256) void pfn1_wmma_kernel(
    const _Float16* __restrict__ pf16, const _Float16* __restrict__ wp,
    const float* __restrict__ g1, const float* __restrict__ b1,
    const int* __restrict__ unq_inv, float* __restrict__ fv, int N)
{
  int Mtiles = (N + 15) / 16;
  int mt = blockIdx.x * 8 + (threadIdx.x >> 5);
  if (mt >= Mtiles) return;
  int lane = threadIdx.x & 31;
  int p = mt * 16 + (lane & 15);
  int koff = (lane >> 4) * 8;
  v16h a = {};
  if (p < N) {
    const v8h lo = *(const v8h*)(pf16 + (size_t)p * 32 + koff);
    const v8h hi = *(const v8h*)(pf16 + (size_t)p * 32 + 16 + koff);
#pragma unroll
    for (int j = 0; j < 8; ++j) { a[j] = lo[j]; a[j + 8] = hi[j]; }
  }
  const v8f vzero = {};
  v8f acc[4];
#pragma unroll
  for (int j = 0; j < 4; ++j) acc[j] = vzero;
#pragma unroll
  for (int j = 0; j < 4; ++j) {
    const v16h bfrag = *(const v16h*)(wp + ((size_t)j * 32 + lane) * 16);
    acc[j] = __builtin_amdgcn_wmma_f32_16x16x32_f16(false, a, false, bfrag,
                                                    (short)0, acc[j], false, false);
  }
  int m0 = (lane >> 4) * 8;
  if (mt * 16 + 16 <= N) {       // full tile: unguarded epilogue
    int piv[8];
#pragma unroll
    for (int r = 0; r < 8; ++r) piv[r] = unq_inv[mt * 16 + m0 + r];
#pragma unroll
    for (int j = 0; j < 4; ++j) {
      int n = j * 16 + (lane & 15);
      float g = g1[n], bb = b1[n];
#pragma unroll
      for (int r = 0; r < 8; ++r) {
        float v = fmaxf(acc[j][r] * g + bb, 0.f);
        atomicMax((int*)&fv[(size_t)piv[r] * 64 + n], __float_as_int(v));
      }
    }
  } else {
#pragma unroll
    for (int j = 0; j < 4; ++j) {
      int n = j * 16 + (lane & 15);
      float g = g1[n], bb = b1[n];
#pragma unroll
      for (int r = 0; r < 8; ++r) {
        int pr = mt * 16 + m0 + r;
        if (pr < N) {
          float v = fmaxf(acc[j][r] * g + bb, 0.f);
          atomicMax((int*)&fv[(size_t)unq_inv[pr] * 64 + n], __float_as_int(v));
        }
      }
    }
  }
}

// scatter pillar features to dense fp16 BEV image + occupancy map
__global__ void scatter_bev_kernel(const float* __restrict__ fv,
                                   const int* __restrict__ unq,
                                   _Float16* __restrict__ img,
                                   float* __restrict__ occ, int M, int Hn)
{
  int m = blockIdx.x * blockDim.x + threadIdx.x;
  if (m >= M) return;
  int b = unq[m * 3], y = unq[m * 3 + 1], x = unq[m * 3 + 2];
  size_t base = ((size_t)b * Hn + y) * Hn + x;
  occ[base] = 1.f;
#pragma unroll
  for (int c = 0; c < 64; ++c)
    img[base * 64 + c] = (_Float16)fv[(size_t)m * 64 + c];
}

// active-site mask: 3x3 max-pool (any active input in receptive field)
__global__ void occ_pool_kernel(const float* __restrict__ in, float* __restrict__ out,
                                int Hin, int Hout, int stride)
{
  int idx = blockIdx.x * blockDim.x + threadIdx.x;
  int total = 2 * Hout * Hout;
  if (idx >= total) return;
  int b = idx / (Hout * Hout);
  int rem = idx % (Hout * Hout);
  int oy = rem / Hout, ox = rem % Hout;
  float mx = 0.f;
  for (int dy = -1; dy <= 1; ++dy)
    for (int dx = -1; dx <= 1; ++dx) {
      int iy = oy * stride + dy, ix = ox * stride + dx;
      if (iy >= 0 && iy < Hin && ix >= 0 && ix < Hin)
        mx = fmaxf(mx, in[((size_t)b * Hin + iy) * Hin + ix]);
    }
  out[idx] = mx > 0.f ? 1.f : 0.f;
}

// ---------------------------------------------------------------------------
// Implicit-GEMM 3x3 conv via WMMA, LDS-staged input halo.
// Interior blocks DMA the halo patch with the Tensor Data Mover (one
// tensor_load_to_lds per block, s_wait_tensorcnt + barrier); border blocks
// take the zero-padded VALU fill path. A-frags from ds_load_b128 (halo in
// LDS => no bounds checks, EXEC all-ones at every WMMA); B-frags are one
// contiguous 32B/lane load from the packed weight buffer (L2-resident).
// Fully unrolled taps x KC x NT2 => dense v_wmma issue.
// ---------------------------------------------------------------------------
template <int CIN, int COUT, int STRIDE>
__global__ __launch_bounds__(256) void conv3x3_wmma_kernel(
    const _Float16* __restrict__ in, const _Float16* __restrict__ wp,
    const float* __restrict__ gamma, const float* __restrict__ beta,
    const float* __restrict__ occ, const _Float16* __restrict__ resid,
    _Float16* __restrict__ out, int Hin, int Hout, int relu)
{
  constexpr int KC  = CIN / 32;   // K chunks of 32 per tap
  constexpr int NT2 = COUT / 32;  // WMMA N-tiles per wave (half of COUT)
  constexpr int PH  = 7 * STRIDE + 3;
  constexpr int PW  = PH;
  constexpr int CH8 = CIN / 8;    // 16-byte chunks per pixel
  __shared__ alignas(16) _Float16 patch[PH * PW * CIN];

  const int b   = blockIdx.z;
  const int oy0 = blockIdx.y * 8;
  const int ox0 = blockIdx.x * 8;
  const int iy0 = oy0 * STRIDE - 1;
  const int ix0 = ox0 * STRIDE - 1;

  const bool interior = (iy0 >= 0) && (ix0 >= 0) &&
                        (iy0 + PH <= Hin) && (ix0 + PW <= Hin);
#if __has_builtin(__builtin_amdgcn_tensor_load_to_lds)
  if (interior) {
    if (threadIdx.x < 32) {   // one wave issues the TDM, waits TENSORcnt==0
      unsigned lds_base = (unsigned)(unsigned long long)(const void*)patch;
      tdm_load_patch(in + (((size_t)b * Hin + iy0) * Hin + ix0) * CIN,
                     lds_base, PW * CIN, PH, Hin * CIN);
    }
  } else
#endif
  {
    // cooperative halo-patch fill (zero padding outside the input)
    for (int ch = threadIdx.x; ch < PH * PW * CH8; ch += 256) {
      int c8 = ch % CH8;
      int pix = ch / CH8;
      int px = pix % PW, py = pix / PW;
      int gy = iy0 + py, gx = ix0 + px;
      v8h v = {};
      if (gy >= 0 && gy < Hin && gx >= 0 && gx < Hin)
        v = *(const v8h*)(in + (((size_t)b * Hin + gy) * Hin + gx) * CIN + c8 * 8);
      *(v8h*)(patch + (size_t)pix * CIN + c8 * 8) = v;
    }
  }
  __syncthreads();

  const int lane  = threadIdx.x & 31;
  const int wave  = threadIdx.x >> 5;
  const int msub  = wave >> 1;  // which 16 pixels of the 8x8 tile
  const int nhalf = wave & 1;   // which half of COUT
  const int idx = msub * 16 + (lane & 15);
  const int pyo = idx >> 3, pxo = idx & 7;
  const int koff = (lane >> 4) * 8;

  const v8f vzero = {};
  v8f acc[NT2];
#pragma unroll
  for (int j = 0; j < NT2; ++j) acc[j] = vzero;

#pragma unroll
  for (int t = 0; t < 9; ++t) {
    // prefetch next tap's weight fragments into L2 (global_prefetch_b8)
    if (t < 8)
      __builtin_prefetch(wp + (((size_t)(nhalf * NT2) * 9 + (t + 1)) * KC) * 512 +
                             (size_t)lane * 16, 0, 3);
    const int ly = pyo * STRIDE + t / 3;
    const int lx = pxo * STRIDE + t % 3;
    const _Float16* prow = patch + ((size_t)ly * PW + lx) * CIN;
#pragma unroll
    for (int kc = 0; kc < KC; ++kc) {
      v16h a;
      const v8h lo = *(const v8h*)(prow + kc * 32 + koff);
      const v8h hi = *(const v8h*)(prow + kc * 32 + 16 + koff);
#pragma unroll
      for (int jj = 0; jj < 8; ++jj) { a[jj] = lo[jj]; a[jj + 8] = hi[jj]; }
#pragma unroll
      for (int j = 0; j < NT2; ++j) {
        const int nt = nhalf * NT2 + j;
        const v16h bfrag = *(const v16h*)(
            wp + (((size_t)nt * 9 + t) * KC + kc) * 512 + (size_t)lane * 16);
        acc[j] = __builtin_amdgcn_wmma_f32_16x16x32_f16(
            false, a, false, bfrag, (short)0, acc[j], false, false);
      }
    }
  }

  // epilogue: BN (+ReLU), occupancy mask, optional residual-add + ReLU
  const int m0 = (lane >> 4) * 8;
  size_t spv[8];
  float occv[8];
#pragma unroll
  for (int r = 0; r < 8; ++r) {
    const int id2 = msub * 16 + m0 + r;
    const int oy = oy0 + (id2 >> 3), ox = ox0 + (id2 & 7);
    spv[r] = ((size_t)b * Hout + oy) * Hout + ox;
    occv[r] = occ[spv[r]];
  }
#pragma unroll
  for (int j = 0; j < NT2; ++j) {
    const int n = (nhalf * NT2 + j) * 16 + (lane & 15);
    const float g = gamma[n], bb = beta[n];
#pragma unroll
    for (int r = 0; r < 8; ++r) {
      float v = acc[j][r] * g + bb;
      if (relu) v = fmaxf(v, 0.f);
      v *= occv[r];
      if (resid) { v += (float)resid[spv[r] * COUT + n]; v = fmaxf(v, 0.f); }
      out[spv[r] * COUT + n] = (_Float16)v;
    }
  }
}

// bilinear gather of final 128-ch map at per-point positions; f32 output
__global__ __launch_bounds__(128) void gather_kernel(
    const float* __restrict__ feat, const int* __restrict__ unq,
    const int* __restrict__ unq_inv, const _Float16* __restrict__ x,
    float* __restrict__ out, int N)
{
  const int Hc = 128, Cc = 128;
  int p = blockIdx.x;
  if (p >= N) return;
  int c = threadIdx.x;
  float px = (feat[(size_t)p * 16 + 0] - PC_MIN) / VOXEL * 0.5f;  // / DS
  float py = (feat[(size_t)p * 16 + 1] - PC_MIN) / VOXEL * 0.5f;
  int b = unq[unq_inv[p] * 3];
  int x0 = (int)floorf(px);
  int y0 = (int)floorf(py);
  int x1 = min(max(x0 + 1, 0), Hc - 1);
  int y1 = min(max(y0 + 1, 0), Hc - 1);
  x0 = min(max(x0, 0), Hc - 1);
  y0 = min(max(y0, 0), Hc - 1);
  float x0f = (float)x0, x1f = (float)x1, y0f = (float)y0, y1f = (float)y1;
  float wa = (x1f - px) * (y1f - py);
  float wb = (x1f - px) * (py - y0f);
  float wc = (px - x0f) * (y1f - py);
  float wd = (px - x0f) * (py - y0f);
  size_t base = (size_t)b * Hc * Hc;
  float Ia = (float)x[(base + (size_t)y0 * Hc + x0) * Cc + c];
  float Ib = (float)x[(base + (size_t)y1 * Hc + x0) * Cc + c];
  float Ic = (float)x[(base + (size_t)y0 * Hc + x1) * Cc + c];
  float Id = (float)x[(base + (size_t)y1 * Hc + x1) * Cc + c];
  out[(size_t)p * Cc + c] = Ia * wa + Ib * wb + Ic * wc + Id * wd;
}

// ---------------------------------------------------------------------------
extern "C" void kernel_launch(void* const* d_in, const int* in_sizes, int n_in,
                              void* d_out, int out_size, void* d_ws, size_t ws_size,
                              hipStream_t stream)
{
  const float* feat   = (const float*)d_in[0];
  const int* unq      = (const int*)d_in[1];
  const int* unq_inv  = (const int*)d_in[2];
  const int N  = in_sizes[0] / 16;
  const int Mp = in_sizes[1] / 3;
  const int Hn = 256, H2 = 128;

  char* ws = (char*)d_ws;
  size_t off = 0;
  auto alloc = [&](size_t bytes) -> void* {
    void* r = ws + off;
    off += (bytes + 255) & ~(size_t)255;
    return r;
  };
  _Float16* pf16 = (_Float16*)alloc((size_t)N * 32 * 2);
  float*    hmax = (float*)alloc((size_t)Mp * 16 * 4);
  float*    fv   = (float*)alloc((size_t)Mp * 64 * 4);
  _Float16* wp_p = (_Float16*)alloc((size_t)4 * 32 * 16 * 2);
  _Float16* img  = (_Float16*)alloc((size_t)2 * Hn * Hn * 64 * 2);
  float*    occ0 = (float*)alloc((size_t)2 * Hn * Hn * 4);
  float*    occ1 = (float*)alloc((size_t)2 * Hn * Hn * 4);
  float*    occ2 = (float*)alloc((size_t)2 * H2 * H2 * 4);
  _Float16* wp1d = (_Float16*)alloc((size_t)4 * 9 * 2 * 32 * 16 * 2);
  _Float16* wp11 = (_Float16*)alloc((size_t)4 * 9 * 2 * 32 * 16 * 2);
  _Float16* wp12 = (_Float16*)alloc((size_t)4 * 9 * 2 * 32 * 16 * 2);
  _Float16* wp2d = (_Float16*)alloc((size_t)8 * 9 * 2 * 32 * 16 * 2);
  _Float16* wp21 = (_Float16*)alloc((size_t)8 * 9 * 4 * 32 * 16 * 2);
  _Float16* wp22 = (_Float16*)alloc((size_t)8 * 9 * 4 * 32 * 16 * 2);
  _Float16* xa   = (_Float16*)alloc((size_t)2 * Hn * Hn * 64 * 2);
  _Float16* ya   = (_Float16*)alloc((size_t)2 * Hn * Hn * 64 * 2);
  _Float16* xb1  = (_Float16*)alloc((size_t)2 * Hn * Hn * 64 * 2);
  _Float16* x2   = (_Float16*)alloc((size_t)2 * H2 * H2 * 128 * 2);
  _Float16* y2b  = (_Float16*)alloc((size_t)2 * H2 * H2 * 128 * 2);
  _Float16* xfin = (_Float16*)alloc((size_t)2 * H2 * H2 * 128 * 2);
  (void)ws_size; (void)n_in; (void)out_size;

  (void)hipMemsetAsync(hmax, 0, (size_t)Mp * 16 * 4, stream);
  (void)hipMemsetAsync(fv,   0, (size_t)Mp * 64 * 4, stream);
  (void)hipMemsetAsync(img,  0, (size_t)2 * Hn * Hn * 64 * 2, stream);
  (void)hipMemsetAsync(occ0, 0, (size_t)2 * Hn * Hn * 4, stream);

  auto packLaunch = [&](const void* w, _Float16* wp, int Cin, int Cout, int taps) {
    int total = (Cout / 16) * taps * (Cin / 32) * 32;
    pack_w_kernel<<<(total + 255) / 256, 256, 0, stream>>>((const float*)w, wp,
                                                           Cin, Cout, taps);
  };
  packLaunch(d_in[7],  wp_p, 32, 64, 1);
  packLaunch(d_in[10], wp1d, 64, 64, 9);
  packLaunch(d_in[13], wp11, 64, 64, 9);
  packLaunch(d_in[16], wp12, 64, 64, 9);
  packLaunch(d_in[19], wp2d, 64, 128, 9);
  packLaunch(d_in[22], wp21, 128, 128, 9);
  packLaunch(d_in[25], wp22, 128, 128, 9);

  pfn0_kernel<<<(N + 255) / 256, 256, 0, stream>>>(
      feat, (const float*)d_in[4], (const float*)d_in[5], (const float*)d_in[6],
      unq_inv, pf16, hmax, N);
  concat_hmax_kernel<<<(N + 255) / 256, 256, 0, stream>>>(pf16, hmax, unq_inv, N);
  {
    int mtiles = (N + 15) / 16;
    pfn1_wmma_kernel<<<(mtiles + 7) / 8, 256, 0, stream>>>(
        pf16, wp_p, (const float*)d_in[8], (const float*)d_in[9], unq_inv, fv, N);
  }
  scatter_bev_kernel<<<(Mp + 255) / 256, 256, 0, stream>>>(fv, unq, img, occ0, Mp, Hn);
  occ_pool_kernel<<<(2 * Hn * Hn + 255) / 256, 256, 0, stream>>>(occ0, occ1, Hn, Hn, 1);
  occ_pool_kernel<<<(2 * H2 * H2 + 255) / 256, 256, 0, stream>>>(occ1, occ2, Hn, H2, 2);

  const dim3 g1(Hn / 8, Hn / 8, 2);   // block-1 convs @ 256x256
  const dim3 g2(H2 / 8, H2 / 8, 2);   // block-2 convs @ 128x128
  conv3x3_wmma_kernel<64, 64, 1><<<g1, 256, 0, stream>>>(
      img, wp1d, (const float*)d_in[11], (const float*)d_in[12], occ1, nullptr,
      xa, Hn, Hn, 1);
  conv3x3_wmma_kernel<64, 64, 1><<<g1, 256, 0, stream>>>(
      xa, wp11, (const float*)d_in[14], (const float*)d_in[15], occ1, nullptr,
      ya, Hn, Hn, 1);
  conv3x3_wmma_kernel<64, 64, 1><<<g1, 256, 0, stream>>>(
      ya, wp12, (const float*)d_in[17], (const float*)d_in[18], occ1, xa,
      xb1, Hn, Hn, 0);
  conv3x3_wmma_kernel<64, 128, 2><<<g2, 256, 0, stream>>>(
      xb1, wp2d, (const float*)d_in[20], (const float*)d_in[21], occ2, nullptr,
      x2, Hn, H2, 1);
  conv3x3_wmma_kernel<128, 128, 1><<<g2, 256, 0, stream>>>(
      x2, wp21, (const float*)d_in[23], (const float*)d_in[24], occ2, nullptr,
      y2b, H2, H2, 1);
  conv3x3_wmma_kernel<128, 128, 1><<<g2, 256, 0, stream>>>(
      y2b, wp22, (const float*)d_in[26], (const float*)d_in[27], occ2, x2,
      xfin, H2, H2, 0);

  gather_kernel<<<N, 128, 0, stream>>>(feat, unq, unq_inv, xfin, (float*)d_out, N);
}
